// ChunkedAttention4D_2937757630660
// MI455X (gfx1250) — matrix operand order repaired and checked by reading, not compiled, not run on hardware
//
#include <hip/hip_runtime.h>
#include <hip/hip_bf16.h>

// B=4, C=8, S=1024, E=512, H=8, D=64.  M = B*C*S = 32768 rows.
// MI455X / gfx1250, wave32.  All GEMMs via v_wmma_f32_16x16x32_bf16.
// Flash-attention online softmax per (head, 16-row q-tile).
// Row-sum of exp(P) computed by an extra WMMA against a ones-tile (saves
// 32 ds_bpermute per key block); only row-max uses cross-lane shuffles.
// Weights pre-converted to bf16 once so the projection hot loop has zero
// scalar converts on the B side.

typedef __bf16 bf16_t;
typedef __attribute__((ext_vector_type(16))) __bf16 v16bf;
typedef __attribute__((ext_vector_type(8)))  __bf16 v8bf;
typedef __attribute__((ext_vector_type(8)))  float  v8f;

__device__ inline v8f wmma_bf16(v16bf a, v16bf b, v8f c) {
  return __builtin_amdgcn_wmma_f32_16x16x32_bf16(false, a, false, b, (short)0, c,
                                                 false, false);
}

// ---------------- WMMA operand loaders (wave32 layouts, CDNA5 ISA) --------
// A 16x32 bf16: lane L holds row (L&15); half-wave hl=(L>>4) selects K base
// hl*8 for VGPRs 0..3 and 16+hl*8 for VGPRs 4..7.
__device__ inline v16bf load_a(const float* p, int ld, int lane) {
  const int row = lane & 15, hl = lane >> 4;
  const float* r = p + row * ld + hl * 8;
  v16bf a;
#pragma unroll
  for (int j = 0; j < 8; ++j) {
    a[j]     = (__bf16)r[j];
    a[8 + j] = (__bf16)r[16 + j];
  }
  return a;
}

__device__ inline v16bf load_a(const bf16_t* p, int ld, int lane) {
  const int row = lane & 15, hl = lane >> 4;
  const bf16_t* r = p + row * ld + hl * 8;
  v8bf lo = *(const v8bf*)r;           // 16B aligned in all uses
  v8bf hi = *(const v8bf*)(r + 16);
  v16bf a;
#pragma unroll
  for (int j = 0; j < 8; ++j) { a[j] = lo[j]; a[8 + j] = hi[j]; }
  return a;
}

// B 32x16 bf16: lane L holds column (L&15); half-wave hl=(L>>4) selects K base
// hl*8 for elements 0..7 and 16+hl*8 for elements 8..15 (same as A).
__device__ inline v16bf load_b(const bf16_t* p, int ld, int lane) {
  const int col = lane & 15, hl = lane >> 4;
  const bf16_t* r = p + col * ld + hl * 8;
  v8bf lo = *(const v8bf*)r;
  v8bf hi = *(const v8bf*)(r + 16);
  v16bf b;
#pragma unroll
  for (int j = 0; j < 8; ++j) { b[j] = lo[j]; b[8 + j] = hi[j]; }
  return b;
}

__device__ inline float rowmax16(float v) {
  v = fmaxf(v, __shfl_xor(v, 1, 32));
  v = fmaxf(v, __shfl_xor(v, 2, 32));
  v = fmaxf(v, __shfl_xor(v, 4, 32));
  v = fmaxf(v, __shfl_xor(v, 8, 32));
  return v;
}

// ---------------- fp32 -> bf16 weight pre-conversion -----------------------
__global__ void cvt_kernel(const float* __restrict__ src, bf16_t* __restrict__ dst,
                           int n) {
  const int i = (blockIdx.x * 256 + threadIdx.x) * 4;
  if (i + 3 < n) {
    const float4 f = *(const float4*)(src + i);
    dst[i]     = (bf16_t)f.x;
    dst[i + 1] = (bf16_t)f.y;
    dst[i + 2] = (bf16_t)f.z;
    dst[i + 3] = (bf16_t)f.w;
  }
}

// ---------------- Projection GEMM:  out = X @ W^T + bias  (NT form) --------
// X: [32768, 512] (AT = float or bf16), Wbf: [512,512] bf16 row-major.
// MODE 0: bf16 out, layout [bc, h, s, d]      (Q, K)
// MODE 1: bf16 out, layout [bc, h, d, s]      (V transposed)
// MODE 2: fp32 out, flat [m, n]               (final O projection)
template <int MODE, typename AT>
__global__ void proj_kernel(const AT* __restrict__ X, const bf16_t* __restrict__ Wbf,
                            const float* __restrict__ bias, void* __restrict__ outp) {
  const int lane = threadIdx.x & 31;
  const int wave = threadIdx.x >> 5;                 // 8 waves / block
  const int mbase = blockIdx.x * 128 + wave * 16;    // 16 rows per wave
  const int nbase = blockIdx.y * 64;                 // 64 cols per block
  const int hl = lane >> 4, ln = lane & 15;

  v8f acc[4] = {};
  for (int k = 0; k < 512; k += 32) {
    v16bf a = load_a(X + (size_t)mbase * 512 + k, 512, lane);
#pragma unroll
    for (int t = 0; t < 4; ++t) {
      v16bf b = load_b(Wbf + (size_t)(nbase + t * 16) * 512 + k, 512, lane);
      acc[t] = wmma_bf16(a, b, acc[t]);
    }
  }
#pragma unroll
  for (int t = 0; t < 4; ++t) {
    const int n = nbase + t * 16 + ln;
    const float bv = bias[n];
#pragma unroll
    for (int r = 0; r < 8; ++r) {
      const int m = mbase + hl * 8 + r;              // C/D: M=(L>>4)*8+r, N=L&15
      const float val = acc[t][r] + bv;
      if (MODE == 2) {
        ((float*)outp)[(size_t)m * 512 + n] = val;
      } else {
        const int bc = m >> 10, s = m & 1023;
        const int h = n >> 6, d = n & 63;
        const size_t idx = (MODE == 0)
            ? ((size_t)((bc * 8 + h) * 1024 + s) * 64 + d)
            : ((size_t)((bc * 8 + h) * 64 + d) * 1024 + s);
        ((bf16_t*)outp)[idx] = (bf16_t)val;
      }
    }
  }
}

// ---------------- Flash attention step: one 32-key block -------------------
template <bool MASKED>
__device__ inline void flash_step(int kb, int qbase, int lane,
                                  const bf16_t* __restrict__ kh,
                                  const bf16_t* __restrict__ vh,
                                  v16bf qa0, v16bf qa1, v16bf ones,
                                  bf16_t* pl, v8f (&acc)[4], v8f& lsum,
                                  float (&mrun)[8]) {
  const int hl = lane >> 4, ln = lane & 15;
  // scores: S0 = cols kb..kb+15, S1 = cols kb+16..kb+31 (contraction d=64)
  v8f s0 = {}, s1 = {};
  {
    v16bf b0 = load_b(kh + (size_t)kb * 64, 64, lane);
    v16bf b1 = load_b(kh + (size_t)kb * 64 + 32, 64, lane);
    s0 = wmma_bf16(qa0, b0, s0);
    s0 = wmma_bf16(qa1, b1, s0);
  }
  {
    v16bf b2 = load_b(kh + (size_t)(kb + 16) * 64, 64, lane);
    v16bf b3 = load_b(kh + (size_t)(kb + 16) * 64 + 32, 64, lane);
    s1 = wmma_bf16(qa0, b2, s1);
    s1 = wmma_bf16(qa1, b3, s1);
  }
  const float scale = 0.125f;                        // 1/sqrt(64)
  const int c0 = kb + ln, c1 = c0 + 16;
  // fused softmax update: scalars only (no live p/alpha arrays -> no spills)
#pragma unroll
  for (int r = 0; r < 8; ++r) {
    float x0 = s0[r] * scale;
    float x1 = s1[r] * scale;
    if (MASKED) {
      const int m = qbase + hl * 8 + r;
      x0 = (c0 <= m) ? x0 : -1e30f;
      x1 = (c1 <= m) ? x1 : -1e30f;
    }
    const float mx = rowmax16(fmaxf(x0, x1));
    const float mnew = fmaxf(mrun[r], mx);
    const float alpha = __expf(mrun[r] - mnew);
    mrun[r] = mnew;
    const float p0 = __expf(x0 - mnew);
    const float p1 = __expf(x1 - mnew);
    acc[0][r] *= alpha; acc[1][r] *= alpha;
    acc[2][r] *= alpha; acc[3][r] *= alpha;
    lsum[r] *= alpha;
    const int ml = hl * 8 + r;
    pl[ml * 32 + ln]      = (bf16_t)p0;
    pl[ml * 32 + ln + 16] = (bf16_t)p1;
  }
  __syncthreads();                                   // single-wave WG: waits only
  const v16bf pa = load_a((const bf16_t*)pl, 32, lane);
  lsum = wmma_bf16(pa, ones, lsum);                  // row-sum via P @ 1 (WMMA)
#pragma unroll
  for (int t = 0; t < 4; ++t) {                      // O += P @ V (contraction=seq)
    v16bf vb = load_b(vh + (size_t)(t * 16) * 1024 + kb, 1024, lane);
    acc[t] = wmma_bf16(pa, vb, acc[t]);
  }
  __syncthreads();
}

// ---------------- Flash attention: one wave per (head, 16-row q-tile) ------
// Q,K: [bc,h,s,d] bf16;  VT: [bc,h,d,s] bf16;  O: [bc,s,e] bf16.
__global__ void __launch_bounds__(32, 1)
attn_kernel(const bf16_t* __restrict__ Q, const bf16_t* __restrict__ K,
            const bf16_t* __restrict__ VT, bf16_t* __restrict__ O) {
  __shared__ __align__(16) bf16_t pl[16 * 32];       // P-tile relayout buffer
  const int lane = threadIdx.x & 31;
  const int hl = lane >> 4, ln = lane & 15;
  const int qt = blockIdx.x & 63;                    // 64 q-tiles of 16 rows
  const int headlin = blockIdx.x >> 6;               // 256 (bc,h)
  const int h = headlin & 7, bc = headlin >> 3;
  const int qbase = qt * 16;

  const bf16_t* qh = Q  + (size_t)(bc * 8 + h) * 1024 * 64;
  const bf16_t* kh = K  + (size_t)(bc * 8 + h) * 1024 * 64;
  const bf16_t* vh = VT + (size_t)(bc * 8 + h) * 64 * 1024;

  const v16bf qa0 = load_a(qh + (size_t)qbase * 64, 64, lane);       // d 0..31
  const v16bf qa1 = load_a(qh + (size_t)qbase * 64 + 32, 64, lane);  // d 32..63
  v16bf ones;
#pragma unroll
  for (int j = 0; j < 16; ++j) ones[j] = (__bf16)1.0f;

  v8f acc[4] = {};
  v8f lsum = {};
  float mrun[8];
#pragma unroll
  for (int r = 0; r < 8; ++r) mrun[r] = -1e30f;

  const int nfull = qbase >> 5;                      // mask-free 32-key blocks
  for (int i = 0; i < nfull; ++i)
    flash_step<false>(i * 32, qbase, lane, kh, vh, qa0, qa1, ones, pl, acc,
                      lsum, mrun);
  flash_step<true>(nfull * 32, qbase, lane, kh, vh, qa0, qa1, ones, pl, acc,
                   lsum, mrun);                      // diagonal block

  // epilogue: divide by row sum, emit [bc, s, e] bf16 (feeds O-projection)
#pragma unroll
  for (int r = 0; r < 8; ++r) {
    const float inv = 1.0f / lsum[r];
    const int m = qbase + hl * 8 + r;
    const size_t base = ((size_t)(bc * 1024 + m)) * 512 + h * 64 + ln;
#pragma unroll
    for (int t = 0; t < 4; ++t)
      O[base + t * 16] = (bf16_t)(acc[t][r] * inv);
  }
}

extern "C" void kernel_launch(void* const* d_in, const int* in_sizes, int n_in,
                              void* d_out, int out_size, void* d_ws, size_t ws_size,
                              hipStream_t stream) {
  (void)in_sizes; (void)n_in; (void)out_size; (void)ws_size;
  const float* q  = (const float*)d_in[0];
  const float* k  = (const float*)d_in[1];
  const float* v  = (const float*)d_in[2];
  const float* Wq = (const float*)d_in[3];
  const float* bq = (const float*)d_in[4];
  const float* Wk = (const float*)d_in[5];
  const float* bk = (const float*)d_in[6];
  const float* Wv = (const float*)d_in[7];
  const float* bv = (const float*)d_in[8];
  const float* Wo = (const float*)d_in[9];
  const float* bo = (const float*)d_in[10];

  const size_t ME = (size_t)32768 * 512;             // 16M elems per tensor
  const size_t WE = (size_t)512 * 512;               // 256K elems per weight
  bf16_t* q_ws  = (bf16_t*)d_ws;                     // [bc,h,s,d]
  bf16_t* k_ws  = q_ws + ME;                         // [bc,h,s,d]
  bf16_t* vT_ws = k_ws + ME;                         // [bc,h,d,s]
  bf16_t* o_ws  = vT_ws + ME;                        // [bc,s,e]
  bf16_t* wq_bf = o_ws + ME;
  bf16_t* wk_bf = wq_bf + WE;
  bf16_t* wv_bf = wk_bf + WE;
  bf16_t* wo_bf = wv_bf + WE;

  // 1) weights fp32 -> bf16 (once; keeps proj hot loop convert-free on B side)
  {
    dim3 cblk(256), cgrd(WE / (256 * 4));
    cvt_kernel<<<cgrd, cblk, 0, stream>>>(Wq, wq_bf, (int)WE);
    cvt_kernel<<<cgrd, cblk, 0, stream>>>(Wk, wk_bf, (int)WE);
    cvt_kernel<<<cgrd, cblk, 0, stream>>>(Wv, wv_bf, (int)WE);
    cvt_kernel<<<cgrd, cblk, 0, stream>>>(Wo, wo_bf, (int)WE);
  }

  dim3 blk(256);                                     // 8 waves
  dim3 grd(32768 / 128, 512 / 64);
  proj_kernel<0, float><<<grd, blk, 0, stream>>>(q, wq_bf, bq, q_ws);
  proj_kernel<0, float><<<grd, blk, 0, stream>>>(k, wk_bf, bk, k_ws);
  proj_kernel<1, float><<<grd, blk, 0, stream>>>(v, wv_bf, bv, vT_ws);
  attn_kernel<<<dim3(256 * 64), dim3(32), 0, stream>>>(q_ws, k_ws, vT_ws, o_ws);
  proj_kernel<2, bf16_t><<<grd, blk, 0, stream>>>(o_ws, wo_bf, bo, d_out);
}
